// SelfAttention3D_65953517797929
// MI455X (gfx1250) — compile-verified, matrix operand order checked by reading, not run on hardware
//
#include <hip/hip_runtime.h>
#include <stdint.h>

#define BATCH 4
#define CCH   128
#define CQ    16
#define NTOK  4096
#define CP    32      // q/k channel dim padded to one bf16-WMMA K
#define WAVES 4

typedef __attribute__((ext_vector_type(16))) __bf16 v16bf;
typedef __attribute__((ext_vector_type(8)))  float  v8f;

// A/B 16-bit WMMA operand: u[0..3] and u[4..7] are each 16 contiguous bytes
union ABReg { uint4 q[2]; v16bf v; };

#if defined(__has_builtin)
#if __has_builtin(__builtin_amdgcn_tensor_load_to_lds)
#define HAVE_TDM 1
#endif
#endif

#ifdef HAVE_TDM
typedef __attribute__((ext_vector_type(4))) unsigned int u32x4;
typedef __attribute__((ext_vector_type(4))) int i32x4;
typedef __attribute__((ext_vector_type(8))) int i32x8;

// Issue one TDM 2-D tile load (bf16 elements): tile0 x tile1 out of a
// dim0 x dim1 tensor with row stride stride0, into LDS at lds_addr.
__device__ __forceinline__ void tdm_load_2d(uint32_t lds_addr, uint64_t gaddr,
                                            unsigned dim0, unsigned dim1,
                                            unsigned stride0,
                                            unsigned tile0, unsigned tile1) {
  u32x4 g0;
  g0.x = 1u;                                                    // count=1, user D#
  g0.y = lds_addr;                                              // LDS byte address
  g0.z = (uint32_t)gaddr;                                       // global_addr[31:0]
  g0.w = (uint32_t)((gaddr >> 32) & 0x01FFFFFFull) | (2u << 30);// addr[56:32] | type=2
  i32x8 g1;
  g1[0] = (int)(1u << 16);                                      // data_size=1 (2B)
  g1[1] = (int)(dim0 << 16);                                    // tensor_dim0 lo16
  g1[2] = (int)((dim0 >> 16) | (dim1 << 16));                   // dim0 hi / dim1 lo
  g1[3] = (int)((dim1 >> 16) | (tile0 << 16));                  // dim1 hi / tile_dim0
  g1[4] = (int)tile1;                                           // tile_dim1 (tile_dim2=0)
  g1[5] = (int)stride0;                                         // tensor_dim0_stride lo32
  g1[6] = 0;
  g1[7] = 0;
  i32x4 z4 = {0, 0, 0, 0};
#if __clang_major__ >= 23
  i32x8 z8 = {0, 0, 0, 0, 0, 0, 0, 0};
  __builtin_amdgcn_tensor_load_to_lds(g0, g1, z4, z4, z8, 0);
#else
  __builtin_amdgcn_tensor_load_to_lds(g0, g1, z4, z4, 0);
#endif
}
#endif

__device__ __forceinline__ unsigned short f2bf(float f) {
  unsigned int u = __float_as_uint(f);
  u += 0x7fffu + ((u >> 16) & 1u);     // round-to-nearest-even
  return (unsigned short)(u >> 16);
}

// ---------------- projection: q,k -> bf16 [B][N][32] (channels 16..31 zero) ----
__global__ void proj_qk_kernel(const float* __restrict__ x,
                               const float* __restrict__ Wq, const float* __restrict__ bq,
                               const float* __restrict__ Wk, const float* __restrict__ bk,
                               unsigned short* __restrict__ qpad,
                               unsigned short* __restrict__ kpad) {
  const int n  = blockIdx.x * blockDim.x + threadIdx.x;
  const int oc = blockIdx.y;   // 0..31 (>=16 are zero padding)
  const int b  = blockIdx.z;
  const size_t outIdx = ((size_t)b * NTOK + n) * CP + oc;
  if (oc >= CQ) { qpad[outIdx] = 0; kpad[outIdx] = 0; return; }
  float aq = bq[oc], ak = bk[oc];
  const float* xb = x + (size_t)b * CCH * NTOK + n;
  #pragma unroll 4
  for (int c = 0; c < CCH; ++c) {
    float xv = xb[(size_t)c * NTOK];
    aq = fmaf(Wq[oc * CCH + c], xv, aq);
    ak = fmaf(Wk[oc * CCH + c], xv, ak);
  }
  qpad[outIdx] = f2bf(aq);
  kpad[outIdx] = f2bf(ak);
}

// ---------------- projection: v -> bf16 vT [B][C][N] (m fastest) ---------------
__global__ void proj_v_kernel(const float* __restrict__ x,
                              const float* __restrict__ Wv, const float* __restrict__ bv,
                              unsigned short* __restrict__ vT) {
  const int n = blockIdx.x * blockDim.x + threadIdx.x;
  const int c = blockIdx.y;
  const int b = blockIdx.z;
  float acc = bv[c];
  const float* xb = x + (size_t)b * CCH * NTOK + n;
  #pragma unroll 4
  for (int ci = 0; ci < CCH; ++ci)
    acc = fmaf(Wv[c * CCH + ci], xb[(size_t)ci * NTOK], acc);
  vT[((size_t)b * CCH + c) * NTOK + n] = f2bf(acc);
}

// ---------------- fused flash attention -----------------------------------
__global__ __launch_bounds__(128) void attn_kernel(
    const unsigned short* __restrict__ qpad,
    const unsigned short* __restrict__ kpad,
    const unsigned short* __restrict__ vT,
    const float* __restrict__ x,
    const float* __restrict__ gamma,
    float* __restrict__ out) {
  __shared__ __align__(16) unsigned short ktile[32 * CP];       // [m=32][c=32]
  __shared__ __align__(16) unsigned short vtile[CCH * 32];      // [c=128][m=32]
  __shared__ __align__(16) unsigned short pb[WAVES][16 * 32];   // P bf16 [n=16][m=32]
  __shared__ __align__(16) float          sbuf[WAVES][16 * 32]; // S f32  [n=16][m=32]
  __shared__ __align__(16) float          scb[WAVES][16];       // per-row broadcast

  const int tid  = threadIdx.x;
  const int lane = tid & 31;
  const int w    = tid >> 5;
  const int r    = lane & 15;   // row / column index within a 16-lane half
  const int h    = lane >> 4;   // which half of the wave
  const int b    = blockIdx.y;
  const int n0   = (blockIdx.x * WAVES + w) * 16;
  const float g  = gamma[0];

  // A operand (Q tile), loaded once: two global b128 per lane
  ABReg aq;
  {
    const uint4* qb = (const uint4*)(qpad + ((size_t)b * NTOK + n0 + r) * CP);
    aq.q[0] = qb[h];
    aq.q[1] = qb[2 + h];
  }

  const v8f vzero = {0.f, 0.f, 0.f, 0.f, 0.f, 0.f, 0.f, 0.f};
  v8f O[8];
  #pragma unroll
  for (int cc = 0; cc < 8; ++cc) O[cc] = vzero;
  float rowM = -3.0e38f, rowL = 0.f;   // softmax state for row r (dup in both halves)

  // LDS byte offsets (generic LDS pointer carries offset in low 32 bits)
  const uint32_t ldsK     = (uint32_t)(uintptr_t)&ktile[0] + (uint32_t)tid * 16u;
  const uint32_t ldsVBase = (uint32_t)(uintptr_t)&vtile[0];

  for (int m0 = 0; m0 < NTOK; m0 += 32) {
    // ---- V tile (128x32 bf16, 8KB) via Tensor Data Mover, issued by wave 0 ----
#ifdef HAVE_TDM
    if (w == 0) {
      tdm_load_2d(ldsVBase,
                  (uint64_t)(uintptr_t)(vT + (size_t)b * CCH * NTOK + m0),
                  /*dim0=*/NTOK, /*dim1=*/CCH, /*stride0=*/NTOK,
                  /*tile0=*/32, /*tile1=*/CCH);
    }
#else
    {
      const unsigned short* gvp = vT + ((size_t)b * CCH + tid) * NTOK + m0;
      #pragma unroll
      for (int i = 0; i < 4; ++i) {
        uint64_t gv = (uint64_t)(uintptr_t)(gvp + i * 8);
        asm volatile("global_load_async_to_lds_b128 %0, %1, off"
                     :: "v"(ldsVBase + (uint32_t)tid * 64u + (uint32_t)(i * 16)),
                        "v"(gv) : "memory");
      }
    }
#endif
    // ---- K tile (32x32 bf16, 2KB) via async global->LDS, all 128 threads ----
    {
      uint64_t gk = (uint64_t)(uintptr_t)(kpad + ((size_t)b * NTOK + m0 + (tid >> 2)) * CP
                                          + (size_t)(tid & 3) * 8);
      asm volatile("global_load_async_to_lds_b128 %0, %1, off" :: "v"(ldsK), "v"(gk) : "memory");
    }
    if (m0 + 32 < NTOK)   // global_prefetch_b8 for the next V tile
      __builtin_prefetch(vT + ((size_t)b * CCH + tid) * NTOK + m0 + 32, 0, 1);
#ifdef HAVE_TDM
    if (w == 0) __builtin_amdgcn_s_wait_tensorcnt(0);
#endif
    asm volatile("s_wait_asynccnt 0" ::: "memory");
    __syncthreads();

    // ---- S = Q^T K : two 16x16 bf16 WMMAs ----
    ABReg bk0, bk1;
    {
      const uint4* kt4 = (const uint4*)ktile;      // 4 uint4 per 32-elem row
      bk0.q[0] = kt4[r * 4 + h];        bk0.q[1] = kt4[r * 4 + 2 + h];
      bk1.q[0] = kt4[(r + 16) * 4 + h]; bk1.q[1] = kt4[(r + 16) * 4 + 2 + h];
    }
    v8f s0 = __builtin_amdgcn_wmma_f32_16x16x32_bf16(false, aq.v, false, bk0.v,
                                                     (short)0, vzero, false, false);
    v8f s1 = __builtin_amdgcn_wmma_f32_16x16x32_bf16(false, aq.v, false, bk1.v,
                                                     (short)0, vzero, false, false);

    // ---- transpose S through LDS: C-layout -> row-major [16][32] ----
    #pragma unroll
    for (int j = 0; j < 8; ++j) {
      int row = j + 8 * h;
      sbuf[w][row * 32 + r]      = s0[j];
      sbuf[w][row * 32 + 16 + r] = s1[j];
    }
    asm volatile("s_wait_dscnt 0" ::: "memory");

    // ---- row-wise online softmax: lane owns row r, halves split the 32 cols ----
    float vls[16];
    {
      const float4* sr = (const float4*)&sbuf[w][r * 32 + h * 16];
      #pragma unroll
      for (int i = 0; i < 4; ++i) {
        float4 f = sr[i];
        vls[4 * i] = f.x; vls[4 * i + 1] = f.y; vls[4 * i + 2] = f.z; vls[4 * i + 3] = f.w;
      }
    }
    float mx = vls[0];
    #pragma unroll
    for (int i = 1; i < 16; ++i) mx = fmaxf(mx, vls[i]);
    mx = fmaxf(mx, __shfl_xor(mx, 16, 32));          // single cross-half combine
    float nm = fmaxf(rowM, mx);
    float sc = __expf(rowM - nm);
    rowM = nm;
    float ls = 0.f;
    unsigned int pk[8];
    #pragma unroll
    for (int i = 0; i < 16; i += 2) {
      float p0 = __expf(vls[i] - nm);
      float p1 = __expf(vls[i + 1] - nm);
      ls += p0 + p1;
      pk[i >> 1] = (unsigned int)f2bf(p0) | ((unsigned int)f2bf(p1) << 16);
    }
    ls += __shfl_xor(ls, 16, 32);
    rowL = rowL * sc + ls;
    {
      unsigned int* pbr = (unsigned int*)&pb[w][r * 32 + h * 16];
      #pragma unroll
      for (int i = 0; i < 8; ++i) pbr[i] = pk[i];
    }
    if (h == 0) scb[w][r] = sc;                      // broadcast rescale factor
    asm volatile("s_wait_dscnt 0" ::: "memory");

    // ---- rescale O (C-layout rows j + 8h) ----
    {
      const float4* sp = (const float4*)&scb[w][8 * h];
      float4 sA = sp[0], sB = sp[1];
      float scl[8] = {sA.x, sA.y, sA.z, sA.w, sB.x, sB.y, sB.z, sB.w};
      #pragma unroll
      for (int cc = 0; cc < 8; ++cc) {
        #pragma unroll
        for (int j = 0; j < 8; ++j) O[cc][j] *= scl[j];
      }
    }

    // ---- O += P x V^T : 8 bf16 WMMAs over 128 value channels ----
    ABReg ap;
    {
      const uint4* pb4 = (const uint4*)&pb[w][0];
      ap.q[0] = pb4[r * 4 + h];
      ap.q[1] = pb4[r * 4 + 2 + h];
    }
    const uint4* vt4 = (const uint4*)vtile;
    #pragma unroll
    for (int cc = 0; cc < 8; ++cc) {
      ABReg bv;
      int crow = cc * 16 + r;
      bv.q[0] = vt4[crow * 4 + h];
      bv.q[1] = vt4[crow * 4 + 2 + h];
      O[cc] = __builtin_amdgcn_wmma_f32_16x16x32_bf16(false, ap.v, false, bv.v,
                                                      (short)0, O[cc], false, false);
    }
    __syncthreads();
  }

  // ---- epilogue: out = gamma * (O / L) + x ----
  if (h == 0) scb[w][r] = 1.0f / rowL;
  asm volatile("s_wait_dscnt 0" ::: "memory");
  float inv[8];
  {
    const float4* ip = (const float4*)&scb[w][8 * h];
    float4 iA = ip[0], iB = ip[1];
    inv[0]=iA.x; inv[1]=iA.y; inv[2]=iA.z; inv[3]=iA.w;
    inv[4]=iB.x; inv[5]=iB.y; inv[6]=iB.z; inv[7]=iB.w;
  }
  #pragma unroll
  for (int cc = 0; cc < 8; ++cc) {
    #pragma unroll
    for (int j = 0; j < 8; ++j) {
      int c = cc * 16 + r;
      int n = n0 + j + 8 * h;
      size_t idx = ((size_t)b * CCH + c) * NTOK + n;
      out[idx] = g * (O[cc][j] * inv[j]) + x[idx];
    }
  }
}

extern "C" void kernel_launch(void* const* d_in, const int* in_sizes, int n_in,
                              void* d_out, int out_size, void* d_ws, size_t ws_size,
                              hipStream_t stream) {
  (void)in_sizes; (void)n_in; (void)out_size; (void)ws_size;
  const float* x     = (const float*)d_in[0];
  const float* Wq    = (const float*)d_in[1];
  const float* bq    = (const float*)d_in[2];
  const float* Wk    = (const float*)d_in[3];
  const float* bk    = (const float*)d_in[4];
  const float* Wv    = (const float*)d_in[5];
  const float* bv    = (const float*)d_in[6];
  const float* gamma = (const float*)d_in[7];
  float* out = (float*)d_out;

  unsigned short* qpad = (unsigned short*)d_ws;                   // 1 MB
  unsigned short* kpad = qpad + (size_t)BATCH * NTOK * CP;        // 1 MB
  unsigned short* vT   = kpad + (size_t)BATCH * NTOK * CP;        // 4 MB

  proj_qk_kernel<<<dim3(NTOK / 256, CP, BATCH), 256, 0, stream>>>(x, Wq, bq, Wk, bk,
                                                                  qpad, kpad);
  proj_v_kernel<<<dim3(NTOK / 256, CCH, BATCH), 256, 0, stream>>>(x, Wv, bv, vT);
  attn_kernel<<<dim3(NTOK / (16 * WAVES), BATCH), 128, 0, stream>>>(qpad, kpad, vT, x,
                                                                    gamma, out);
}
